// CrossModalAttentionFusion_14018773254229
// MI455X (gfx1250) — compile-verified, matrix operand order checked by reading, not run on hardware
//
#include <hip/hip_runtime.h>

#define DDIM 1024
#define BROWS 16384

typedef __bf16 bf16;
typedef __attribute__((ext_vector_type(16))) bf16 v16bf;
typedef __attribute__((ext_vector_type(8)))  bf16 v8bf;
typedef __attribute__((ext_vector_type(4)))  bf16 v4bf;
typedef __attribute__((ext_vector_type(8)))  float v8f;

// ---------------------------------------------------------------------------
// fp32 -> bf16 weight conversion (one-shot per weight matrix, D*D elements)
// ---------------------------------------------------------------------------
__global__ __launch_bounds__(256) void cvt_f32_bf16_kernel(const float* __restrict__ src,
                                                           bf16* __restrict__ dst) {
    int i = (blockIdx.x * 256 + threadIdx.x) * 4;
    float4 f = *(const float4*)(src + i);
    v4bf o;
    o[0] = (bf16)f.x; o[1] = (bf16)f.y; o[2] = (bf16)f.z; o[3] = (bf16)f.w;
    *(v4bf*)(dst + i) = o;
}

// ---------------------------------------------------------------------------
// Fragment load from an LDS tile laid out [row][32] bf16, following the
// 16-bit 16x32 A-matrix layout (B symmetric with lane index = N):
// lane l<16 holds row (base+l), K = {0..7, 16..23}; lane>=16: K = {8..15, 24..31}.
// ---------------------------------------------------------------------------
__device__ __forceinline__ v16bf load_frag(const bf16* tile, int row_base, int lane) {
    const bf16* p = tile + (unsigned)(row_base + (lane & 15)) * 32 + ((lane >> 4) * 8);
    v8bf c0 = *(const v8bf*)(p);        // 16B ds_load
    v8bf c1 = *(const v8bf*)(p + 16);   // 16B ds_load
    return __builtin_shufflevector(c0, c1, 0,1,2,3,4,5,6,7,8,9,10,11,12,13,14,15);
}

__device__ __forceinline__ v8bf cvt8(float4 a, float4 b) {
    v8bf r;
    r[0] = (bf16)a.x; r[1] = (bf16)a.y; r[2] = (bf16)a.z; r[3] = (bf16)a.w;
    r[4] = (bf16)b.x; r[5] = (bf16)b.y; r[6] = (bf16)b.z; r[7] = (bf16)b.w;
    return r;
}

// ---------------------------------------------------------------------------
// gfx1250 async global->LDS copy, 16 bytes per lane, tracked by ASYNCcnt.
// (GLOBAL_LOAD_ASYNC_TO_LDS_B128, ISA 15.18.3 op 98.)  Generic LDS pointer's
// low 32 bits are the LDS byte offset (VDST operand).
// ---------------------------------------------------------------------------
__device__ __forceinline__ void async_copy_b128(bf16* lds_dst, const bf16* gsrc) {
    unsigned ldsoff = (unsigned)(unsigned long long)lds_dst;
    asm volatile("global_load_async_to_lds_b128 %0, %1, off"
                 :: "v"(ldsoff), "v"(gsrc) : "memory");
}
__device__ __forceinline__ void wait_asynccnt0() {
    asm volatile("s_wait_asynccnt 0" ::: "memory");
}

// ---------------------------------------------------------------------------
// C[b, j] = sum_d (A[b,d] (+ A2[b,d])) * W[j,d] + bias[j]
//   A, A2 : fp32 row-major [BROWS, D]  (converted to bf16 in-flight)
//   W     : bf16 row-major [D, D]      (pre-converted; staged via async-to-LDS)
// 256x128 C-tile per 256-thread block; 8 waves in 4(M) x 2(N), each wave owns a
// 64x64 sub-tile (16 accumulators). K staged in 32-wide slabs through
// double-buffered LDS (A: global->reg->LDS with cvt; W: async global->LDS).
// ---------------------------------------------------------------------------
template <bool HAS_A2>
__global__ __launch_bounds__(256) void gemm_nt_bf16_wmma_kernel(
        const float* __restrict__ A, const float* __restrict__ A2,
        const bf16* __restrict__ W, const float* __restrict__ bias,
        float* __restrict__ C) {
    __shared__ alignas(16) bf16 sA[2][256 * 32];   // 2 x 16 KB
    __shared__ alignas(16) bf16 sW[2][128 * 32];   // 2 x  8 KB

    const int tid  = threadIdx.x;
    const int lane = tid & 31;
    const int wave = tid >> 5;
    const int wm   = wave >> 1;        // 0..3 -> 64-row (M) slice
    const int wn   = wave & 1;         // 0..1 -> 64-col (N) slice
    const int row0 = blockIdx.y * 256;
    const int col0 = blockIdx.x * 128;

    // staging coords: A covers 256 rows x 32 cols via 2 tasks/thread,
    // W covers 128 rows x 32 cols via 1 task/thread (16 elems per task)
    const int ar0 = tid >> 1;            // 0..127
    const int ar1 = ar0 + 128;           // 128..255
    const int ac  = (tid & 1) * 16;      // 0 or 16
    const int wr  = tid >> 1;            // 0..127
    const int wc  = (tid & 1) * 16;      // 0 or 16

    const float* pa0 = A + (size_t)(row0 + ar0) * DDIM + ac;
    const float* pa1 = A + (size_t)(row0 + ar1) * DDIM + ac;
    const float* pb0 = HAS_A2 ? A2 + (size_t)(row0 + ar0) * DDIM + ac : nullptr;
    const float* pb1 = HAS_A2 ? A2 + (size_t)(row0 + ar1) * DDIM + ac : nullptr;
    const bf16*  pw  = W + (size_t)(col0 + wr) * DDIM + wc;

    float4 fa[8];

    // ---- A: global -> registers for K slab at offset k (uniform k) ----
    auto g2rA = [&](int k) {
        const float4* qa0 = (const float4*)(pa0 + k);
        const float4* qa1 = (const float4*)(pa1 + k);
        fa[0] = qa0[0]; fa[1] = qa0[1]; fa[2] = qa0[2]; fa[3] = qa0[3];
        fa[4] = qa1[0]; fa[5] = qa1[1]; fa[6] = qa1[2]; fa[7] = qa1[3];
        if (HAS_A2) {
            const float4* qb0 = (const float4*)(pb0 + k);
            const float4* qb1 = (const float4*)(pb1 + k);
            #pragma unroll
            for (int i = 0; i < 4; ++i) {
                float4 g = qb0[i];
                fa[i].x += g.x; fa[i].y += g.y; fa[i].z += g.z; fa[i].w += g.w;
            }
            #pragma unroll
            for (int i = 0; i < 4; ++i) {
                float4 g = qb1[i];
                fa[4+i].x += g.x; fa[4+i].y += g.y; fa[4+i].z += g.z; fa[4+i].w += g.w;
            }
        }
        __builtin_prefetch(pa0 + k + 32, 0, 0);   // speculative next-slab hint
    };

    // ---- W: async global -> LDS for K slab at offset k into buffer b ----
    auto asyncW = [&](int k, int b) {
        bf16* dst = &sW[b][wr * 32 + wc];
        async_copy_b128(dst,     pw + k);
        async_copy_b128(dst + 8, pw + k + 8);
    };

    v8f acc[4][4] = {};

    asyncW(0, 0);
    g2rA(0);
    int buf = 0;
    for (int k0 = 0; k0 < DDIM; k0 += 32) {
        // ---- A registers -> LDS (fp32 -> bf16) ----
        {
            bf16* da0 = &sA[buf][ar0 * 32 + ac];
            *(v8bf*)(da0)     = cvt8(fa[0], fa[1]);
            *(v8bf*)(da0 + 8) = cvt8(fa[2], fa[3]);
            bf16* da1 = &sA[buf][ar1 * 32 + ac];
            *(v8bf*)(da1)     = cvt8(fa[4], fa[5]);
            *(v8bf*)(da1 + 8) = cvt8(fa[6], fa[7]);
        }
        wait_asynccnt0();      // this wave's W(buf) lanes have landed in LDS
        __syncthreads();       // ... and everyone else's

        if (k0 + 32 < DDIM) {  // overlap next slab with WMMA
            asyncW(k0 + 32, buf ^ 1);
            g2rA(k0 + 32);
        }

        v16bf bfrag[4];
        #pragma unroll
        for (int ni = 0; ni < 4; ++ni)
            bfrag[ni] = load_frag(sW[buf], wn * 64 + ni * 16, lane);

        #pragma unroll
        for (int mi = 0; mi < 4; ++mi) {
            v16bf afrag = load_frag(sA[buf], wm * 64 + mi * 16, lane);
            #pragma unroll
            for (int ni = 0; ni < 4; ++ni) {
                acc[mi][ni] = __builtin_amdgcn_wmma_f32_16x16x32_bf16(
                    false, afrag, false, bfrag[ni],
                    (short)0, acc[mi][ni], false, false);
            }
        }
        buf ^= 1;
    }

    // ---- epilogue: 16x16 f32 C layout: VGPR r -> row r + 8*(lane>=16),
    //      col = lane & 15 ----
    float bv[4];
    #pragma unroll
    for (int ni = 0; ni < 4; ++ni)
        bv[ni] = bias[col0 + wn * 64 + ni * 16 + (lane & 15)];

    #pragma unroll
    for (int mi = 0; mi < 4; ++mi) {
        const int grbase = row0 + wm * 64 + mi * 16 + ((lane >> 4) * 8);
        #pragma unroll
        for (int ni = 0; ni < 4; ++ni) {
            const int gc = col0 + wn * 64 + ni * 16 + (lane & 15);
            #pragma unroll
            for (int r = 0; r < 8; ++r)
                C[(size_t)(grbase + r) * DDIM + gc] = acc[mi][ni][r] + bv[ni];
        }
    }
}

// ---------------------------------------------------------------------------
// Host launcher.  Reference collapses (softmax over a length-1 axis == 1) to:
//   fused     = (text+image) @ fuse_w.T + fuse_b
//   out_text  = (fused @ Wv_t.T + bv_t) @ t_out_w.T + t_out_b
//   out_image = (fused @ Wv_i.T + bv_i) @ i_out_w.T + i_out_b
// d_out = [out_text | out_image | fused], each BROWS*D fp32.
// ---------------------------------------------------------------------------
extern "C" void kernel_launch(void* const* d_in, const int* in_sizes, int n_in,
                              void* d_out, int out_size, void* d_ws, size_t ws_size,
                              hipStream_t stream) {
    const float* text    = (const float*)d_in[0];
    const float* image   = (const float*)d_in[1];
    const float* fuse_w  = (const float*)d_in[2];
    const float* fuse_b  = (const float*)d_in[3];
    const float* t_in_w  = (const float*)d_in[4];
    const float* t_in_b  = (const float*)d_in[5];
    const float* t_out_w = (const float*)d_in[6];
    const float* t_out_b = (const float*)d_in[7];
    const float* i_in_w  = (const float*)d_in[8];
    const float* i_in_b  = (const float*)d_in[9];
    const float* i_out_w = (const float*)d_in[10];
    const float* i_out_b = (const float*)d_in[11];

    const size_t BD = (size_t)BROWS * DDIM;
    const size_t DD = (size_t)DDIM * DDIM;

    float* out_text  = (float*)d_out;
    float* out_image = out_text + BD;
    float* fused     = out_image + BD;

    // workspace: 5 bf16 [D,D] weights (10 MB), then one fp32 [B,D] temp (64 MB)
    bf16* w_fuse = (bf16*)d_ws;
    bf16* w_vt   = w_fuse + DD;
    bf16* w_to   = w_vt   + DD;
    bf16* w_vi   = w_to   + DD;
    bf16* w_io   = w_vi   + DD;
    float* tmp   = (float*)(w_io + DD);

    const int cvt_blocks = (int)(DD / 4 / 256);   // 1024
    cvt_f32_bf16_kernel<<<cvt_blocks, 256, 0, stream>>>(fuse_w,          w_fuse);
    cvt_f32_bf16_kernel<<<cvt_blocks, 256, 0, stream>>>(t_in_w + 2 * DD, w_vt);
    cvt_f32_bf16_kernel<<<cvt_blocks, 256, 0, stream>>>(t_out_w,         w_to);
    cvt_f32_bf16_kernel<<<cvt_blocks, 256, 0, stream>>>(i_in_w + 2 * DD, w_vi);
    cvt_f32_bf16_kernel<<<cvt_blocks, 256, 0, stream>>>(i_out_w,         w_io);

    dim3 grid(DDIM / 128, BROWS / 256);   // (8, 64)
    // 1) fused = (text + image) @ fuse_w.T + fuse_b
    gemm_nt_bf16_wmma_kernel<true ><<<grid, 256, 0, stream>>>(text,  image,  w_fuse, fuse_b,            fused);
    // 2) tmp = fused @ Wv_t.T + bv_t ; out_text = tmp @ t_out_w.T + t_out_b
    gemm_nt_bf16_wmma_kernel<false><<<grid, 256, 0, stream>>>(fused, nullptr, w_vt,  t_in_b + 2 * DDIM, tmp);
    gemm_nt_bf16_wmma_kernel<false><<<grid, 256, 0, stream>>>(tmp,   nullptr, w_to,  t_out_b,           out_text);
    // 3) tmp = fused @ Wv_i.T + bv_i ; out_image = tmp @ i_out_w.T + i_out_b
    gemm_nt_bf16_wmma_kernel<false><<<grid, 256, 0, stream>>>(fused, nullptr, w_vi,  i_in_b + 2 * DDIM, tmp);
    gemm_nt_bf16_wmma_kernel<false><<<grid, 256, 0, stream>>>(tmp,   nullptr, w_io,  i_out_b,           out_image);
}
